// UserModel_14654428414525
// MI455X (gfx1250) — matrix operand order repaired and checked by reading, not compiled, not run on hardware
//
#include <hip/hip_runtime.h>
#include <hip/hip_bf16.h>

#define B_   32
#define S_   100
#define T_   4
#define NC2_ 500
#define NC3_ 2001   // NUM_C3 + 1
#define DV_  128

typedef __attribute__((ext_vector_type(16))) _Float16 v16h;
typedef __attribute__((ext_vector_type(8)))  float    v8f;

#define WMMA_F16(A, Bm, C) \
  __builtin_amdgcn_wmma_f32_16x16x32_f16(false, (A), false, (Bm), (short)0, (C), false, false)

// Fast transcendental paths: v_exp_f32 + v_rcp_f32, no IEEE divide chain.
__device__ __forceinline__ float fsigmoid(float x) {
  return __builtin_amdgcn_rcpf(1.f + __expf(-x));
}
__device__ __forceinline__ float ftanh(float x) {
  return 1.f - 2.f * __builtin_amdgcn_rcpf(__expf(2.f * x) + 1.f);
}

// ---------------------------------------------------------------------------
// A-fragment (16x32 f16, ISA 7.12.2 layout) gathered from a row-major matrix
// (LDS-staged activations). lane<16 holds M=lane, K-halves {0..7,16..23};
// lane>=16 holds same M, K-halves {8..15,24..31}.
// ---------------------------------------------------------------------------
__device__ __forceinline__ v16h ldA(const _Float16* M, int ld, int mrow, int kbase) {
  const int lane = threadIdx.x & 31;
  const _Float16* p = M + (size_t)(mrow + (lane & 15)) * ld + kbase + ((lane >> 4) << 3);
  v16h a;
#pragma unroll
  for (int h = 0; h < 16; ++h) a[h] = p[(h & 7) + ((h >> 3) << 4)];
  return a;
}

// B-fragment (32x16 f16): pre-swizzled so each lane reads 16 contiguous halves.
__device__ __forceinline__ v16h ldB(const _Float16* Bp, int nt, int kc, int KC) {
  const int lane = threadIdx.x & 31;
  const _Float16* p = Bp + ((((size_t)nt * KC + kc) * 32 + lane) << 4);
  v16h b;
#pragma unroll
  for (int h = 0; h < 16; ++h) b[h] = p[h];
  return b;
}

// Wave32 lane reduction (5 shuffles).
__device__ __forceinline__ float wave_sum(float v) {
#pragma unroll
  for (int off = 16; off > 0; off >>= 1) v += __shfl_down(v, off, 32);
  return v;
}

// ---------------------------------------------------------------------------
// Pack kernel: W is [Nout][Ktot] f32 (torch Linear weight, used as x @ W.T).
// dst[((nt*KC+kc)*32+lane)*16+h] = W[nt*16+(lane&15)][kc*32 + (h&7) + ((h>>3)*16) + ((lane>>4)*8)]
// ---------------------------------------------------------------------------
__global__ void pack_b_kernel(const float* __restrict__ src, _Float16* __restrict__ dst,
                              int NT, int KC, int Ktot) {
  int t = blockIdx.x * blockDim.x + threadIdx.x;
  if (t >= NT * KC * 512) return;
  int h    = t & 15;
  int lane = (t >> 4) & 31;
  int rem  = t >> 9;
  int kc   = rem % KC;
  int nt   = rem / KC;
  int row  = nt * 16 + (lane & 15);
  int colk = kc * 32 + (h & 7) + ((h >> 3) << 4) + ((lane >> 4) << 3);
  dst[t] = (_Float16)src[(size_t)row * Ktot + colk];
}

// x16[b,s,0:128] = D_emb[d_seq[b,s]] * v_d ;  x16[b,s,128:256] = R_emb[r_seq[b,s]]
__global__ void build_x16_kernel(const int* __restrict__ d_seq, const int* __restrict__ r_seq,
                                 const float* __restrict__ D_emb, const float* __restrict__ R_emb,
                                 const float* __restrict__ v_d, _Float16* __restrict__ x16) {
  int t = blockIdx.x * blockDim.x + threadIdx.x;
  if (t >= B_ * S_ * 256) return;
  int c  = t & 255;
  int bs = t >> 8;
  float v;
  if (c < 128) v = D_emb[d_seq[bs]] * v_d[c];
  else         v = R_emb[r_seq[bs] * DV_ + (c - 128)];
  x16[t] = (_Float16)v;
}

// ---------------------------------------------------------------------------
// GRU: single persistent workgroup, 16 waves, LDS-resident hidden state.
// Gates laid out [r(128) | z(128) | n(128)]. r/z fuse x- and h-matmuls into one
// accumulator; n keeps i_n / h_n separate (n = tanh(i_n + r*h_n)).
// 48 16x16 output-tile jobs per step, 3 per wave, 12 WMMAs per job.
// ---------------------------------------------------------------------------
__global__ __launch_bounds__(512) void gru_kernel(
    const _Float16* __restrict__ x16, const _Float16* __restrict__ WihP,
    const _Float16* __restrict__ WhhP, const float* __restrict__ b_ih,
    const float* __restrict__ b_hh, float* __restrict__ h_out) {
  __shared__ _Float16 xbuf[32 * 256];   // 16 KB current-step input
  __shared__ _Float16 hbuf[32 * 128];   //  8 KB hidden state (f16)
  __shared__ _Float16 rzbuf[32 * 256];  // 16 KB sigmoid(r), sigmoid(z)
  __shared__ _Float16 inbuf[32 * 128];  //  8 KB i_n + b_ih
  __shared__ _Float16 hnbuf[32 * 128];  //  8 KB h_n + b_hh
  const int tid  = threadIdx.x;
  const int lane = tid & 31;
  const int wv   = tid >> 5;

  for (int i = tid; i < 32 * 128; i += 512) hbuf[i] = (_Float16)0.f;
  __syncthreads();

  for (int s = 0; s < S_; ++s) {
    // stage x[:, s, :] into LDS (each thread: 16 contiguous halves)
    {
      int b = tid >> 4, ch = tid & 15;
      const _Float16* src = x16 + ((size_t)b * S_ + s) * 256 + ch * 16;
      _Float16* dst = xbuf + b * 256 + ch * 16;
#pragma unroll
      for (int j = 0; j < 16; ++j) dst[j] = src[j];
      if (s + 1 < S_) __builtin_prefetch((const void*)(src + 256), 0, 1);
    }
    __syncthreads();

#pragma unroll
    for (int jj = 0; jj < 3; ++jj) {
      int job = wv + jj * 16;
      if (job < 32) {  // r/z gates: fully fused x + h matmul, then sigmoid
        int mt = job >> 4, nt = job & 15;
        v8f acc = {};
        for (int kc = 0; kc < 8; ++kc)
          acc = WMMA_F16(ldA(xbuf, 256, mt * 16, kc * 32), ldB(WihP, nt, kc, 8), acc);
        for (int kc = 0; kc < 4; ++kc)
          acc = WMMA_F16(ldA(hbuf, 128, mt * 16, kc * 32), ldB(WhhP, nt, kc, 4), acc);
        int col = nt * 16 + (lane & 15);
        float bsum = b_ih[col] + b_hh[col];
#pragma unroll
        for (int i = 0; i < 8; ++i) {
          int row = mt * 16 + i + ((lane >> 4) << 3);
          rzbuf[row * 256 + col] = (_Float16)fsigmoid(acc[i] + bsum);
        }
      } else {         // n gate: keep input / hidden parts separate
        int j2 = job - 32;
        int mt = j2 >> 3, nt = j2 & 7;
        v8f ai = {}, ah = {};
        for (int kc = 0; kc < 8; ++kc)
          ai = WMMA_F16(ldA(xbuf, 256, mt * 16, kc * 32), ldB(WihP, 16 + nt, kc, 8), ai);
        for (int kc = 0; kc < 4; ++kc)
          ah = WMMA_F16(ldA(hbuf, 128, mt * 16, kc * 32), ldB(WhhP, 16 + nt, kc, 4), ah);
        int c = nt * 16 + (lane & 15);
        float bi = b_ih[256 + c], bh = b_hh[256 + c];
#pragma unroll
        for (int i = 0; i < 8; ++i) {
          int row = mt * 16 + i + ((lane >> 4) << 3);
          inbuf[row * 128 + c] = (_Float16)(ai[i] + bi);
          hnbuf[row * 128 + c] = (_Float16)(ah[i] + bh);
        }
      }
    }
    __syncthreads();

    // elementwise GRU update: 8 CONTIGUOUS elements per thread -> coalesced
    // LDS reads and vector-width global stores of h_seq.
    {
      int base = tid * 8;                 // 512 threads * 8 = 4096 = 32*128
      int m = base >> 7, c0 = base & 127;
      float* hdst = h_out + ((size_t)m * S_ + s) * DV_ + c0;
#pragma unroll
      for (int j = 0; j < 8; ++j) {
        int c = c0 + j, e = base + j;
        float r = (float)rzbuf[m * 256 + c];
        float z = (float)rzbuf[m * 256 + 128 + c];
        float n = ftanh((float)inbuf[e] + r * (float)hnbuf[e]);
        float hnew = (1.f - z) * n + z * (float)hbuf[e];
        hbuf[e] = (_Float16)hnew;
        hdst[j] = hnew;
      }
    }
    __syncthreads();
  }
}

// ---------------------------------------------------------------------------
// alpha = relu(h @ l1_W1.T + b1) @ l1_W2.T + b2 ; one 16-row tile per block.
// ---------------------------------------------------------------------------
__global__ __launch_bounds__(128) void alpha_kernel(
    const float* __restrict__ h_seq, const _Float16* __restrict__ L1P,
    const float* __restrict__ b1, const float* __restrict__ W2,
    const float* __restrict__ b2, float* __restrict__ out_alpha) {
  __shared__ _Float16 ht[16 * 128];
  __shared__ float hid[16 * 128];
  const int tid = threadIdx.x, lane = tid & 31, wv = tid >> 5;
  const size_t row0 = (size_t)blockIdx.x * 16;

  for (int e = tid; e < 16 * 128; e += 128) ht[e] = (_Float16)h_seq[row0 * DV_ + e];
  __syncthreads();

#pragma unroll
  for (int p = 0; p < 2; ++p) {
    int nt = wv + p * 4;
    v8f acc = {};
    for (int kc = 0; kc < 4; ++kc)
      acc = WMMA_F16(ldA(ht, 128, 0, kc * 32), ldB(L1P, nt, kc, 4), acc);
    int col = nt * 16 + (lane & 15);
    float bc = b1[col];
#pragma unroll
    for (int i = 0; i < 8; ++i)
      hid[(i + ((lane >> 4) << 3)) * 128 + col] = fmaxf(acc[i] + bc, 0.f);
  }
  __syncthreads();

  // lane-parallel 128->1 head: 4 rows per wave, shuffle reduction.
#pragma unroll
  for (int rr = 0; rr < 4; ++rr) {
    int m = wv * 4 + rr;
    float acc = 0.f;
#pragma unroll
    for (int c = lane; c < 128; c += 32) acc += hid[m * 128 + c] * W2[c];
    acc = wave_sum(acc);
    if (lane == 0) out_alpha[row0 + m] = acc + b2[0];
  }
}

// ---------------------------------------------------------------------------
// Cache scan: one block per batch. C2/C3 caches live in LDS for all 100 steps.
// Per step: build a 16x512 feature tile (row 0 = l2 input, rows 1..4 = l3
// inputs for the T tags), run both 512->128 matmuls sharing A fragments,
// reduce the 128->1 heads, scatter-update the caches, stream the snapshot.
// ---------------------------------------------------------------------------
__global__ __launch_bounds__(256) void cache_kernel(
    const int* __restrict__ c2_seq, const int* __restrict__ c3_seq,
    const _Float16* __restrict__ x16, const _Float16* __restrict__ L2P,
    const _Float16* __restrict__ L3P, const float* __restrict__ v_c2,
    const float* __restrict__ v_c3, const float* __restrict__ l2_b1,
    const float* __restrict__ l2_W2, const float* __restrict__ l2_b2,
    const float* __restrict__ l3_b1, const float* __restrict__ l3_W2,
    const float* __restrict__ l3_b2, float* __restrict__ out_C2,
    float* __restrict__ out_C3) {
  __shared__ float    C2c[NC2_];        //  2.0 KB
  __shared__ float    C3c[NC3_];        //  8.0 KB
  __shared__ _Float16 feat[16 * 512];   // 16.0 KB
  __shared__ float    hid2[16 * 128];   //  8.0 KB
  __shared__ float    hid3[16 * 128];   //  8.0 KB
  __shared__ float sc_beta2, sc_b3bar, sc_beta3[T_], sc_new[5];
  __shared__ int   sc_idx3[T_], sc_mask[T_], sc_c2;

  const int tid = threadIdx.x, lane = tid & 31, wv = tid >> 5;
  const int b = blockIdx.x;

  for (int i = tid; i < NC2_; i += 256) C2c[i] = 0.f;
  for (int i = tid; i < NC3_; i += 256) C3c[i] = 0.f;
  __syncthreads();

  for (int s = 0; s < S_; ++s) {
    if (tid == 0) {  // betas read from the carry (pre-update) caches
      sc_c2 = c2_seq[b * S_ + s];
      float msum = 0.f;
      for (int t = 0; t < T_; ++t) {
        int cv = c3_seq[(b * S_ + s) * T_ + t];
        sc_idx3[t] = cv;
        sc_mask[t] = (cv != 0);
        msum += (float)sc_mask[t];
      }
      sc_beta2 = C2c[sc_c2];
      float inv = (msum > 0.f) ? __builtin_amdgcn_rcpf(msum) : 0.f, bb = 0.f;
      for (int t = 0; t < T_; ++t) {
        float bt = C3c[sc_idx3[t]];
        sc_beta3[t] = bt;
        if (sc_mask[t]) bb += bt * inv;
      }
      sc_b3bar = bb;
    }
    __syncthreads();

    // feature tile: [vc2 | vc3 | vd | vr], rows >=5 zero-padded
    for (int e = tid; e < 16 * 512; e += 256) {
      int r = e >> 9, c = e & 511;
      _Float16 v = (_Float16)0.f;
      if (r < 5) {
        if (c < 128)      v = (_Float16)(sc_beta2 * v_c2[c]);
        else if (c < 256) {
          float coef = (r == 0) ? sc_b3bar : sc_beta3[r - 1];
          v = (_Float16)(coef * v_c3[c - 128]);
        } else            v = x16[((size_t)b * S_ + s) * 256 + (c - 256)];
      }
      feat[r * 512 + c] = v;
    }
    __syncthreads();

    if (wv < 8) {  // shared A fragment feeds both l2 and l3 WMMAs
      v8f a2 = {}, a3 = {};
      for (int kc = 0; kc < 16; ++kc) {
        v16h af = ldA(feat, 512, 0, kc * 32);
        a2 = WMMA_F16(af, ldB(L2P, wv, kc, 16), a2);
        a3 = WMMA_F16(af, ldB(L3P, wv, kc, 16), a3);
      }
      int col = wv * 16 + (lane & 15);
      float bb2 = l2_b1[col], bb3 = l3_b1[col];
#pragma unroll
      for (int i = 0; i < 8; ++i) {
        int row = i + ((lane >> 4) << 3);
        hid2[row * 128 + col] = fmaxf(a2[i] + bb2, 0.f);
        hid3[row * 128 + col] = fmaxf(a3[i] + bb3, 0.f);
      }
    }
    __syncthreads();

    // 128->1 heads: one wave per output row, lane-parallel + shuffle reduce.
    if (wv < 5) {
      const float* w  = (wv == 0) ? l2_W2 : l3_W2;
      const float* hr = (wv == 0) ? &hid2[0] : &hid3[wv * 128];
      float acc = 0.f;
#pragma unroll
      for (int c = lane; c < 128; c += 32) acc += hr[c] * w[c];
      acc = wave_sum(acc);
      if (lane == 0) sc_new[wv] = acc + ((wv == 0) ? l2_b2[0] : l3_b2[0]);
    }
    __syncthreads();

    if (tid == 0) {  // scatter update, matching reference multi-hot semantics
      C2c[sc_c2] = sc_new[0];
      float oldv[T_], res[T_];
      for (int t = 0; t < T_; ++t) oldv[t] = C3c[sc_idx3[t]];
      for (int t = 0; t < T_; ++t) {
        if (!sc_mask[t]) continue;
        float cnt = 0.f, sm = 0.f;
        for (int u = 0; u < T_; ++u)
          if (sc_mask[u] && sc_idx3[u] == sc_idx3[t]) { cnt += 1.f; sm += sc_new[u + 1]; }
        res[t] = oldv[t] * (1.f - cnt) + sm;
      }
      for (int t = 0; t < T_; ++t)
        if (sc_mask[t]) C3c[sc_idx3[t]] = res[t];
    }
    __syncthreads();

    // stream snapshot of both caches (lane-contiguous -> coalesced b32 bursts)
    float* oC2 = out_C2 + ((size_t)b * S_ + s) * NC2_;
    float* oC3 = out_C3 + ((size_t)b * S_ + s) * NC3_;
    for (int i = tid; i < NC2_; i += 256) oC2[i] = C2c[i];
    for (int i = tid; i < NC3_; i += 256) oC3[i] = C3c[i];
    __syncthreads();
  }
}

// ---------------------------------------------------------------------------
extern "C" void kernel_launch(void* const* d_in, const int* in_sizes, int n_in,
                              void* d_out, int out_size, void* d_ws, size_t ws_size,
                              hipStream_t stream) {
  (void)in_sizes; (void)n_in; (void)out_size; (void)ws_size;
  const int*   c2_seq = (const int*)d_in[0];
  const int*   c3_seq = (const int*)d_in[1];
  const int*   d_seq  = (const int*)d_in[2];
  const int*   r_seq  = (const int*)d_in[3];
  const float* D_emb  = (const float*)d_in[4];
  const float* R_emb  = (const float*)d_in[5];
  const float* v_c2   = (const float*)d_in[6];
  const float* v_c3   = (const float*)d_in[7];
  const float* v_d    = (const float*)d_in[8];
  const float* W_ih   = (const float*)d_in[9];
  const float* W_hh   = (const float*)d_in[10];
  const float* b_ih   = (const float*)d_in[11];
  const float* b_hh   = (const float*)d_in[12];
  const float* l1_W1  = (const float*)d_in[13];
  const float* l1_b1  = (const float*)d_in[14];
  const float* l1_W2  = (const float*)d_in[15];
  const float* l1_b2  = (const float*)d_in[16];
  const float* l2_W1  = (const float*)d_in[17];
  const float* l2_b1  = (const float*)d_in[18];
  const float* l2_W2  = (const float*)d_in[19];
  const float* l2_b2  = (const float*)d_in[20];
  const float* l3_W1  = (const float*)d_in[21];
  const float* l3_b1  = (const float*)d_in[22];
  const float* l3_W2  = (const float*)d_in[23];
  const float* l3_b2  = (const float*)d_in[24];

  char* ws = (char*)d_ws;
  _Float16* x16  = (_Float16*)(ws);            // 32*100*256 halves = 1,638,400 B
  _Float16* WihP = (_Float16*)(ws + 1638400);  // 24*8*512 halves   =   196,608 B
  _Float16* WhhP = (_Float16*)(ws + 1835008);  // 24*4*512 halves   =    98,304 B
  _Float16* L1P  = (_Float16*)(ws + 1933312);  //  8*4*512 halves   =    32,768 B
  _Float16* L2P  = (_Float16*)(ws + 1966080);  //  8*16*512 halves  =   131,072 B
  _Float16* L3P  = (_Float16*)(ws + 2097152);  //  8*16*512 halves  =   131,072 B

  float* out       = (float*)d_out;
  float* out_alpha = out;                        // [B,S]        3,200
  float* out_h     = out + 3200;                 // [B,S,DV]   409,600
  float* out_C2    = out_h + 409600;             // [B,S,500] 1,600,000
  float* out_C3    = out_C2 + 1600000;           // [B,S,2001] 6,403,200

  build_x16_kernel<<<(B_ * S_ * 256 + 255) / 256, 256, 0, stream>>>(
      d_seq, r_seq, D_emb, R_emb, v_d, x16);
  pack_b_kernel<<<(24 * 8 * 512 + 255) / 256, 256, 0, stream>>>(W_ih, WihP, 24, 8, 256);
  pack_b_kernel<<<(24 * 4 * 512 + 255) / 256, 256, 0, stream>>>(W_hh, WhhP, 24, 4, 128);
  pack_b_kernel<<<( 8 * 4 * 512 + 255) / 256, 256, 0, stream>>>(l1_W1, L1P, 8, 4, 128);
  pack_b_kernel<<<( 8 * 16 * 512 + 255) / 256, 256, 0, stream>>>(l2_W1, L2P, 8, 16, 512);
  pack_b_kernel<<<( 8 * 16 * 512 + 255) / 256, 256, 0, stream>>>(l3_W1, L3P, 8, 16, 512);

  gru_kernel<<<1, 512, 0, stream>>>(x16, WihP, WhhP, b_ih, b_hh, out_h);
  alpha_kernel<<<200, 128, 0, stream>>>(out_h, L1P, l1_b1, l1_W2, l1_b2, out_alpha);
  cache_kernel<<<32, 256, 0, stream>>>(c2_seq, c3_seq, x16, L2P, L3P, v_c2, v_c3,
                                       l2_b1, l2_W2, l2_b2, l3_b1, l3_W2, l3_b2,
                                       out_C2, out_C3);
}
